// Net_53807350284778
// MI455X (gfx1250) — 8-bit matrix operand order repaired and checked by reading, compiled, not run on hardware
//
#include <hip/hip_runtime.h>

typedef __attribute__((ext_vector_type(2))) float v2f;
typedef __attribute__((ext_vector_type(4))) float v4f;
typedef __attribute__((ext_vector_type(8))) float v8f;
typedef __attribute__((ext_vector_type(4))) int   v4i;
typedef __attribute__((ext_vector_type(8))) int   v8i;
typedef unsigned char u8;
typedef unsigned int  u32;

#if __has_builtin(__builtin_amdgcn_global_load_async_to_lds_b128) && \
    __has_builtin(__builtin_amdgcn_s_wait_asynccnt)
#define HAVE_ASYNC_LDS 1
typedef __attribute__((address_space(1))) v4i* gv4i_p;
typedef __attribute__((address_space(3))) v4i* sv4i_p;
#else
#define HAVE_ASYNC_LDS 0
#endif

__global__ void k_edges(const int* __restrict__ ei, u8* __restrict__ A, int E, int ld) {
  int e = blockIdx.x * blockDim.x + threadIdx.x;
  if (e >= E) return;
  int src = ei[e];
  int dst = ei[E + e];
  A[(size_t)dst * ld + src] = (u8)1;
}

__global__ void k_diag(u8* __restrict__ A, int n, int ld) {
  int i = blockIdx.x * blockDim.x + threadIdx.x;
  if (i < n) A[(size_t)i * ld + i] = (u8)1;
}

__global__ void k_degree(const u8* __restrict__ A, float* __restrict__ dis,
                         int nrows, int ncols, int ld) {
  int row = blockIdx.x;
  int lane = threadIdx.x;
  if (row >= nrows) return;
  const u8* p = A + (size_t)row * ld;
  float ssum = 0.f;
  for (int j = lane; j < ncols; j += 32) ssum += (float)p[j];
  for (int o = 16; o > 0; o >>= 1) ssum += __shfl_xor(ssum, o, 32);
  if (lane == 0) dis[row] = (ssum > 0.f) ? rsqrtf(ssum) : 0.f;
}

__global__ void k_gemm_f32x4(const float* __restrict__ X, const float* __restrict__ W,
                             float* __restrict__ C,
                             int Mtiles, int K, int ldx, int ldw, int ldc) {
  int wave = (blockIdx.x * blockDim.x + threadIdx.x) >> 5;
  int lane = threadIdx.x & 31;
  if (wave >= Mtiles) return;
  int m0 = wave * 16;
  int hh  = lane >> 4;
  int l16 = lane & 15;
  int row = m0 + l16;
  v8f c0 = {0.f,0.f,0.f,0.f,0.f,0.f,0.f,0.f};
  v8f c1 = c0, c2 = c0, c3 = c0;
  const float* xr = X + (size_t)row * ldx + 2 * hh;
  const float* bp = W + (size_t)(2 * hh) * ldw + l16;
  for (int k = 0; k < K; k += 4, xr += 4, bp += 4 * (size_t)ldw) {
    v2f a;
    a.x = xr[0];
    a.y = xr[1];
    v2f b;
    b.x = bp[0];      b.y = bp[ldw];
    c0 = __builtin_amdgcn_wmma_f32_16x16x4_f32(false, a, false, b, (short)0, c0, false, false);
    b.x = bp[16];     b.y = bp[ldw + 16];
    c1 = __builtin_amdgcn_wmma_f32_16x16x4_f32(false, a, false, b, (short)0, c1, false, false);
    b.x = bp[32];     b.y = bp[ldw + 32];
    c2 = __builtin_amdgcn_wmma_f32_16x16x4_f32(false, a, false, b, (short)0, c2, false, false);
    b.x = bp[48];     b.y = bp[ldw + 48];
    c3 = __builtin_amdgcn_wmma_f32_16x16x4_f32(false, a, false, b, (short)0, c3, false, false);
  }
  int rbase = m0 + 8 * hh;
#pragma unroll
  for (int g = 0; g < 8; ++g) {
    float* cr = C + (size_t)(rbase + g) * ldc + l16;
    cr[0]  = c0[g];
    cr[16] = c1[g];
    cr[32] = c2[g];
    cr[48] = c3[g];
  }
}

__global__ void __launch_bounds__(256)
k_gemm_bin64_lds(const u8* __restrict__ A, const float* __restrict__ B,
                 const float* __restrict__ dis, const float* __restrict__ bias,
                 float* __restrict__ C,
                 int Mtiles, int Kn, int lda, int ldb, int ldc, int do_relu) {
  __shared__ float Bs[2][64 * 64];
  int tid  = threadIdx.x;
  int wv   = tid >> 5;
  int lane = tid & 31;
  int mt = blockIdx.x * 8 + wv;
  if (mt >= Mtiles) mt = Mtiles - 1;
  int m0  = mt * 16;
  int hh  = lane >> 4;
  int l16 = lane & 15;
  int row = m0 + l16;
  v8f c0 = {0.f,0.f,0.f,0.f,0.f,0.f,0.f,0.f};
  v8f c1 = c0, c2 = c0, c3 = c0;
  const u8* arow = A + (size_t)row * lda;

  auto stage = [&](int k0, int buf) {
#pragma unroll
    for (int i = 0; i < 4; ++i) {
      int q  = tid * 4 + i;
      int kk = q >> 4;
      int col = (q & 15) * 4;
      const float* g = B + (size_t)(k0 + kk) * ldb + col;
#if HAVE_ASYNC_LDS
      __builtin_amdgcn_global_load_async_to_lds_b128(
          (gv4i_p)(v4i*)g,
          (sv4i_p)(v4i*)&Bs[buf][q * 4], 0, 0);
#else
      *(v4f*)&Bs[buf][q * 4] = *(const v4f*)g;
#endif
    }
  };
  auto compute = [&](const float* bs, const u8* ak) {
#pragma unroll
    for (int kk = 0; kk < 64; kk += 4) {
      int ka = kk + 2 * hh;
      unsigned short aw = *(const unsigned short*)(ak + ka);
      v2f a;
      a.x = (float)(aw & 0xff);
      a.y = (float)(aw >> 8);
      const float* b0r = bs + ka * 64;
      v2f b;
      b.x = b0r[l16];      b.y = b0r[64 + l16];
      c0 = __builtin_amdgcn_wmma_f32_16x16x4_f32(false, a, false, b, (short)0, c0, false, false);
      b.x = b0r[16 + l16]; b.y = b0r[80 + l16];
      c1 = __builtin_amdgcn_wmma_f32_16x16x4_f32(false, a, false, b, (short)0, c1, false, false);
      b.x = b0r[32 + l16]; b.y = b0r[96 + l16];
      c2 = __builtin_amdgcn_wmma_f32_16x16x4_f32(false, a, false, b, (short)0, c2, false, false);
      b.x = b0r[48 + l16]; b.y = b0r[112 + l16];
      c3 = __builtin_amdgcn_wmma_f32_16x16x4_f32(false, a, false, b, (short)0, c3, false, false);
    }
  };

  int nch = Kn >> 6;
  stage(0, 0);
  int c = 0;
  for (; c < nch - 1; ++c) {
    stage((c + 1) << 6, (c & 1) ^ 1);
#if HAVE_ASYNC_LDS
    __builtin_amdgcn_s_wait_asynccnt(4);
#endif
    __syncthreads();
    compute(Bs[c & 1], arow + (c << 6));
    __syncthreads();
  }
#if HAVE_ASYNC_LDS
  __builtin_amdgcn_s_wait_asynccnt(0);
#endif
  __syncthreads();
  compute(Bs[c & 1], arow + (c << 6));

  int rbase = m0 + 8 * hh;
#pragma unroll
  for (int g = 0; g < 8; ++g) {
    float dr = dis[rbase + g];
    float* cr = C + (size_t)(rbase + g) * ldc;
    float v0 = c0[g] * dr + bias[l16];
    float v1 = c1[g] * dr + bias[16 + l16];
    float v2 = c2[g] * dr + bias[32 + l16];
    float v3 = c3[g] * dr + bias[48 + l16];
    if (do_relu) {
      v0 = fmaxf(v0, 0.f); v1 = fmaxf(v1, 0.f);
      v2 = fmaxf(v2, 0.f); v3 = fmaxf(v3, 0.f);
    }
    cr[l16] = v0; cr[16 + l16] = v1; cr[32 + l16] = v2; cr[48 + l16] = v3;
  }
}

__global__ void k_gemm_f32g(const float* __restrict__ X, const float* __restrict__ W,
                            const float* __restrict__ bias, float* __restrict__ C,
                            int Mtiles, int Ntiles, int K,
                            int ldx, int ldw, int ldc, int Nreal) {
  int wave = (blockIdx.x * blockDim.x + threadIdx.x) >> 5;
  int lane = threadIdx.x & 31;
  if (wave >= Mtiles * Ntiles) return;
  int tm = wave / Ntiles, tn = wave - tm * Ntiles;
  int m0 = tm * 16, n0 = tn * 16;
  int hh  = lane >> 4;
  int l16 = lane & 15;
  int row = m0 + l16;
  int col = n0 + l16;
  v8f c = {0.f,0.f,0.f,0.f,0.f,0.f,0.f,0.f};
  const float* xr = X + (size_t)row * ldx + 2 * hh;
  const float* wp = W + (size_t)(2 * hh) * ldw + col;
  for (int k = 0; k < K; k += 4, xr += 4, wp += 4 * (size_t)ldw) {
    v2f a, b;
    a.x = xr[0];
    a.y = xr[1];
    b.x = wp[0];
    b.y = wp[ldw];
    c = __builtin_amdgcn_wmma_f32_16x16x4_f32(false, a, false, b, (short)0, c, false, false);
  }
  if (col < Nreal) {
    int rbase = m0 + 8 * hh;
    float bc = bias[col];
#pragma unroll
    for (int g = 0; g < 8; ++g)
      C[(size_t)(rbase + g) * ldc + col] = c[g] + bc;
  }
}

__device__ __forceinline__ v8i ld_afrag(const u8* p) {
  const u32* q = (const u32*)p;
  v8i a;
  a[0] = (int)q[0];  a[1] = (int)q[1];
  a[2] = (int)q[4];  a[3] = (int)q[5];
  a[4] = (int)q[8];  a[5] = (int)q[9];
  a[6] = (int)q[12]; a[7] = (int)q[13];
  return a;
}
__device__ __forceinline__ v8i ld_bfrag(const u8* p) {
  const u32* q = (const u32*)p;
  v8i b;
  b[0] = (int)q[0]; b[1] = (int)q[1]; b[2] = (int)q[4];  b[3] = (int)q[5];
  b[4] = (int)q[8]; b[5] = (int)q[9]; b[6] = (int)q[12]; b[7] = (int)q[13];
  return b;
}
__global__ void k_augment32(const u8* __restrict__ A, const u8* __restrict__ AT,
                            u8* __restrict__ Out, int T2, int ld) {
  int wave = (blockIdx.x * blockDim.x + threadIdx.x) >> 5;
  int lane = threadIdx.x & 31;
  if (wave >= T2 * T2) return;
  int tm = wave / T2, tn = wave - tm * T2;
  int m0 = tm * 32, n0 = tn * 32;
  int hh  = lane >> 4;
  int l16 = lane & 15;
  v8i z = {0,0,0,0,0,0,0,0};
  v8i c00 = z, c01 = z, c10 = z, c11 = z;
  const u8* pa0 = A  + (size_t)(m0      + l16) * ld + 8 * hh;
  const u8* pa1 = A  + (size_t)(m0 + 16 + l16) * ld + 8 * hh;
  const u8* pb0 = AT + (size_t)(n0      + l16) * ld + 8 * hh;
  const u8* pb1 = AT + (size_t)(n0 + 16 + l16) * ld + 8 * hh;
  for (int k = 0; k < ld; k += 64) {
    __builtin_prefetch(pa0 + k + 256, 0, 1);
    __builtin_prefetch(pb0 + k + 256, 0, 1);
    v8i a0 = ld_afrag(pa0 + k);
    v8i a1 = ld_afrag(pa1 + k);
    v8i b0 = ld_bfrag(pb0 + k);
    v8i b1 = ld_bfrag(pb1 + k);
    c00 = __builtin_amdgcn_wmma_i32_16x16x64_iu8(false, a0, false, b0, c00, false, false);
    c01 = __builtin_amdgcn_wmma_i32_16x16x64_iu8(false, a0, false, b1, c01, false, false);
    c10 = __builtin_amdgcn_wmma_i32_16x16x64_iu8(false, a1, false, b0, c10, false, false);
    c11 = __builtin_amdgcn_wmma_i32_16x16x64_iu8(false, a1, false, b1, c11, false, false);
  }
  int r0 = m0 + 8 * hh;
#pragma unroll
  for (int g = 0; g < 8; ++g) {
    u8* o0 = Out + (size_t)(r0 + g) * ld;
    u8* o1 = Out + (size_t)(r0 + 16 + g) * ld;
    o0[n0 + l16]      = (c00[g] > 0) ? (u8)1 : (u8)0;
    o0[n0 + 16 + l16] = (c01[g] > 0) ? (u8)1 : (u8)0;
    o1[n0 + l16]      = (c10[g] > 0) ? (u8)1 : (u8)0;
    o1[n0 + 16 + l16] = (c11[g] > 0) ? (u8)1 : (u8)0;
  }
}

__global__ void k_scale_rows(float* __restrict__ X, const float* __restrict__ dis,
                             int rows, int ld, int cols) {
  int t = blockIdx.x * blockDim.x + threadIdx.x;
  if (t >= rows * cols) return;
  int r = t / cols, c = t - r * cols;
  X[(size_t)r * ld + c] *= dis[r];
}

__global__ void k_wpad(const float* __restrict__ W, float* __restrict__ Wp,
                       int rows, int cols, int ldp) {
  int t = blockIdx.x * blockDim.x + threadIdx.x;
  if (t >= rows * cols) return;
  int r = t / cols, c = t - r * cols;
  Wp[(size_t)r * ldp + c] = W[(size_t)r * cols + c];
}

__global__ void k_mv64(const float* __restrict__ X, const float* __restrict__ w,
                       float* __restrict__ y, int n, int ldx) {
  int i = blockIdx.x * blockDim.x + threadIdx.x;
  if (i >= n) return;
  float s = 0.f;
#pragma unroll
  for (int c = 0; c < 64; ++c) s += X[(size_t)i * ldx + c] * w[c];
  y[i] = s;
}

__global__ void k_score(const u8* __restrict__ A, const float* __restrict__ y,
                        const float* __restrict__ X, const float* __restrict__ wroot,
                        const float* __restrict__ brel, float* __restrict__ s,
                        int n, int lda, int ldx) {
  int row = blockIdx.x;
  int lane = threadIdx.x;
  if (row >= n) return;
  const u8* ar = A + (size_t)row * lda;
  float acc = 0.f;
  for (int j = lane; j < n; j += 32) acc += ar[j] ? y[j] : 0.f;
  for (int c = lane; c < 64; c += 32) acc += X[(size_t)row * ldx + c] * wroot[c];
  for (int o = 16; o > 0; o >>= 1) acc += __shfl_xor(acc, o, 32);
  if (lane == 0) s[row] = acc + brel[0];
}

__global__ void k_rank(const float* __restrict__ s, float* __restrict__ vals,
                       int* __restrict__ idx, int n, int K) {
  int i = blockIdx.x;
  int lane = threadIdx.x;
  if (i >= n) return;
  float si = s[i];
  int cnt = 0;
  for (int j = lane; j < n; j += 32) {
    float sj = s[j];
    cnt += (sj > si) || (sj == si && j < i);
  }
  for (int o = 16; o > 0; o >>= 1) cnt += __shfl_xor(cnt, o, 32);
  if (lane == 0 && cnt < K) { idx[cnt] = i; vals[cnt] = si; }
}

__global__ void k_gate(const float* __restrict__ X, const int* __restrict__ idx,
                       const float* __restrict__ vals, float* __restrict__ Xp,
                       int K, int ldx) {
  int t = blockIdx.x * blockDim.x + threadIdx.x;
  if (t >= K * 64) return;
  int r = t >> 6, c = t & 63;
  Xp[(size_t)r * 64 + c] = X[(size_t)idx[r] * ldx + c] * tanhf(vals[r]);
}

__global__ void k_scatter_rows(const float* __restrict__ Z, const int* __restrict__ idx,
                               float* __restrict__ Up, int K) {
  int t = blockIdx.x * blockDim.x + threadIdx.x;
  if (t >= K * 64) return;
  int r = t >> 6, c = t & 63;
  Up[(size_t)idx[r] * 64 + c] = Z[(size_t)r * 64 + c];
}

__global__ void k_gather_adj(const u8* __restrict__ A, const int* __restrict__ idx,
                             u8* __restrict__ B, u8* __restrict__ BT,
                             int K, int lda, int ldb) {
  long long t = (long long)blockIdx.x * blockDim.x + threadIdx.x;
  if (t >= (long long)K * K) return;
  int r = (int)(t / K), c = (int)(t - (long long)r * K);
  u8 v = A[(size_t)idx[r] * lda + idx[c]];
  B[(size_t)r * ldb + c]  = v;
  BT[(size_t)c * ldb + r] = v;
}

extern "C" void kernel_launch(void* const* d_in, const int* in_sizes, int n_in,
                              void* d_out, int out_size, void* d_ws, size_t ws_size,
                              hipStream_t stream) {
  (void)in_sizes; (void)n_in; (void)out_size;
  const int N = 4096, E = 65536, FIN = 500;
  const int K1 = 3277, K2 = 2622, K3 = 2098;
  const int LD0 = 4096, LD1 = 3328, LD2 = 2624;
  const int T0 = LD0 / 16, T1 = LD1 / 16, T2 = LD2 / 16;

  const float* x       = (const float*)d_in[0];
  const float* w1      = (const float*)d_in[1];
  const float* b1      = (const float*)d_in[2];
  const float* w2      = (const float*)d_in[3];
  const float* b2      = (const float*)d_in[4];
  const float* w3      = (const float*)d_in[5];
  const float* b3      = (const float*)d_in[6];
  const float* p1_wrel = (const float*)d_in[7];
  const float* p1_brel = (const float*)d_in[8];
  const float* p1_wroot= (const float*)d_in[9];
  const float* p2_wrel = (const float*)d_in[10];
  const float* p2_brel = (const float*)d_in[11];
  const float* p2_wroot= (const float*)d_in[12];
  const float* p3_wrel = (const float*)d_in[13];
  const float* p3_brel = (const float*)d_in[14];
  const float* p3_wroot= (const float*)d_in[15];
  const float* u0_w    = (const float*)d_in[16];
  const float* u0_b    = (const float*)d_in[17];
  const float* u1_w    = (const float*)d_in[18];
  const float* u1_b    = (const float*)d_in[19];
  const float* u2_w    = (const float*)d_in[20];
  const float* u2_b    = (const float*)d_in[21];
  const int*   ei      = (const int*)d_in[22];

  char* base = (char*)d_ws;
  size_t off = 0;
  auto alloc = [&](size_t bytes) { size_t o = off; off = (off + bytes + 255) & ~(size_t)255; return o; };
  u8* A0  = (u8*)(base + alloc((size_t)LD0 * LD0));
  u8* A1  = (u8*)(base + alloc((size_t)LD1 * LD1));
  u8* A1T = (u8*)(base + alloc((size_t)LD1 * LD1));
  u8* A1a = (u8*)(base + alloc((size_t)LD1 * LD1));
  u8* A2  = (u8*)(base + alloc((size_t)LD2 * LD2));
  u8* A2T = (u8*)(base + alloc((size_t)LD2 * LD2));
  u8* A2a = (u8*)(base + alloc((size_t)LD2 * LD2));
  float* xw   = (float*)(base + alloc((size_t)LD0 * 512 * 4));
  float* x1   = (float*)(base + alloc((size_t)LD0 * 64 * 4));
  float* x1p  = (float*)(base + alloc((size_t)LD1 * 64 * 4));
  float* x2   = (float*)(base + alloc((size_t)LD1 * 64 * 4));
  float* x2p  = (float*)(base + alloc((size_t)LD2 * 64 * 4));
  float* x3   = (float*)(base + alloc((size_t)LD2 * 64 * 4));
  float* x3p  = (float*)(base + alloc((size_t)2112 * 64 * 4));
  float* up0  = (float*)(base + alloc((size_t)LD2 * 64 * 4));
  float* z0   = (float*)(base + alloc((size_t)LD2 * 64 * 4));
  float* up1  = (float*)(base + alloc((size_t)LD1 * 64 * 4));
  float* z1   = (float*)(base + alloc((size_t)LD1 * 64 * 4));
  float* up2  = (float*)(base + alloc((size_t)LD0 * 64 * 4));
  float* tbuf = (float*)(base + alloc((size_t)LD0 * 64 * 4));
  float* wpad = (float*)(base + alloc((size_t)64 * 512 * 4));
  float* zb   = (float*)(base + alloc((size_t)512 * 4));
  float* dis0 = (float*)(base + alloc((size_t)LD0 * 4));
  float* dis1 = (float*)(base + alloc((size_t)LD1 * 4));
  float* dis2 = (float*)(base + alloc((size_t)LD2 * 4));
  float* yv   = (float*)(base + alloc((size_t)LD0 * 4));
  float* sv   = (float*)(base + alloc((size_t)LD0 * 4));
  float* vals = (float*)(base + alloc((size_t)LD0 * 4));
  int*  idx1  = (int*)(base + alloc((size_t)LD0 * 4));
  int*  idx2  = (int*)(base + alloc((size_t)LD0 * 4));
  int*  idx3  = (int*)(base + alloc((size_t)LD0 * 4));
  size_t total = off;
  if (total > ws_size) total = ws_size;
  (void)hipMemsetAsync(d_ws, 0, total, stream);

  auto gemmX = [&](const float* X, const float* W, float* C, int Mt, int K,
                   int ldx, int ldw, int ldc) {
    int blocks = (Mt * 32 + 255) / 256;
    k_gemm_f32x4<<<blocks, 256, 0, stream>>>(X, W, C, Mt, K, ldx, ldw, ldc);
  };
  auto gemmB = [&](const u8* A, const float* B, const float* dis, const float* bias,
                   float* C, int Mt, int Kn, int lda, int ldb, int ldc, int relu) {
    int blocks = (Mt + 7) / 8;
    k_gemm_bin64_lds<<<blocks, 256, 0, stream>>>(A, B, dis, bias, C, Mt, Kn, lda, ldb, ldc, relu);
  };

  k_edges<<<(E + 255) / 256, 256, 0, stream>>>(ei, A0, E, LD0);
  k_diag<<<(N + 255) / 256, 256, 0, stream>>>(A0, N, LD0);
  k_degree<<<LD0, 32, 0, stream>>>(A0, dis0, N, N, LD0);

  gemmX(x, w1, xw, T0, FIN, FIN, 64, 512);
  k_scale_rows<<<(N * 64 + 255) / 256, 256, 0, stream>>>(xw, dis0, N, 512, 64);
  gemmB(A0, xw, dis0, b1, x1, T0, N, LD0, 512, 64, 0);

  k_mv64<<<(N + 255) / 256, 256, 0, stream>>>(x1, p1_wrel, yv, N, 64);
  k_score<<<N, 32, 0, stream>>>(A0, yv, x1, p1_wroot, p1_brel, sv, N, LD0, 64);
  k_rank<<<N, 32, 0, stream>>>(sv, vals, idx1, N, K1);
  k_gate<<<(K1 * 64 + 255) / 256, 256, 0, stream>>>(x1, idx1, vals, x1p, K1, 64);
  k_gather_adj<<<(int)(((long long)K1 * K1 + 255) / 256), 256, 0, stream>>>(A0, idx1, A1, A1T, K1, LD0, LD1);

  k_augment32<<<((T1/2) * (T1/2) * 32 + 255) / 256, 256, 0, stream>>>(A1, A1T, A1a, T1 / 2, LD1);
  k_degree<<<LD1, 32, 0, stream>>>(A1a, dis1, LD1, LD1, LD1);

  gemmX(x1p, w2, xw, T1, 64, 64, 64, 512);
  k_scale_rows<<<(LD1 * 64 + 255) / 256, 256, 0, stream>>>(xw, dis1, LD1, 512, 64);
  gemmB(A1a, xw, dis1, b2, x2, T1, LD1, LD1, 512, 64, 0);

  k_mv64<<<(K1 + 255) / 256, 256, 0, stream>>>(x2, p2_wrel, yv, K1, 64);
  k_score<<<K1, 32, 0, stream>>>(A1a, yv, x2, p2_wroot, p2_brel, sv, K1, LD1, 64);
  k_rank<<<K1, 32, 0, stream>>>(sv, vals, idx2, K1, K2);
  k_gate<<<(K2 * 64 + 255) / 256, 256, 0, stream>>>(x2, idx2, vals, x2p, K2, 64);
  k_gather_adj<<<(int)(((long long)K2 * K2 + 255) / 256), 256, 0, stream>>>(A1a, idx2, A2, A2T, K2, LD1, LD2);

  k_augment32<<<((T2/2) * (T2/2) * 32 + 255) / 256, 256, 0, stream>>>(A2, A2T, A2a, T2 / 2, LD2);
  k_degree<<<LD2, 32, 0, stream>>>(A2a, dis2, LD2, LD2, LD2);

  gemmX(x2p, w3, xw, T2, 64, 64, 64, 512);
  k_scale_rows<<<(LD2 * 64 + 255) / 256, 256, 0, stream>>>(xw, dis2, LD2, 512, 64);
  gemmB(A2a, xw, dis2, b3, x3, T2, LD2, LD2, 512, 64, 0);

  k_mv64<<<(K2 + 255) / 256, 256, 0, stream>>>(x3, p3_wrel, yv, K2, 64);
  k_score<<<K2, 32, 0, stream>>>(A2a, yv, x3, p3_wroot, p3_brel, sv, K2, LD2, 64);
  k_rank<<<K2, 32, 0, stream>>>(sv, vals, idx3, K2, K3);
  k_gate<<<(K3 * 64 + 255) / 256, 256, 0, stream>>>(x3, idx3, vals, x3p, K3, 64);

  k_scatter_rows<<<(K3 * 64 + 255) / 256, 256, 0, stream>>>(x3p, idx3, up0, K3);
  gemmX(up0, u0_w, xw, T2, 64, 64, 64, 512);
  k_scale_rows<<<(LD2 * 64 + 255) / 256, 256, 0, stream>>>(xw, dis2, LD2, 512, 64);
  gemmB(A2a, xw, dis2, u0_b, z0, T2, LD2, LD2, 512, 64, 1);

  k_scatter_rows<<<(K2 * 64 + 255) / 256, 256, 0, stream>>>(z0, idx2, up1, K2);
  gemmX(up1, u1_w, xw, T1, 64, 64, 64, 512);
  k_scale_rows<<<(LD1 * 64 + 255) / 256, 256, 0, stream>>>(xw, dis1, LD1, 512, 64);
  gemmB(A1a, xw, dis1, u1_b, z1, T1, LD1, LD1, 512, 64, 1);

  k_scatter_rows<<<(K1 * 64 + 255) / 256, 256, 0, stream>>>(z1, idx1, up2, K1);
  k_scale_rows<<<(N * 64 + 255) / 256, 256, 0, stream>>>(up2, dis0, N, 64, 64);
  gemmB(A0, up2, dis0, zb, tbuf, T0, N, LD0, 64, 64, 0);
  k_wpad<<<(64 * FIN + 255) / 256, 256, 0, stream>>>(u2_w, wpad, 64, FIN, 512);
  {
    int waves = T0 * 32;
    k_gemm_f32g<<<(waves * 32 + 255) / 256, 256, 0, stream>>>(
        tbuf, wpad, u2_b, (float*)d_out, T0, 32, 64, 64, 512, FIN, FIN);
  }
}
